// EvoformerGraphDecoderV2_43885975830847
// MI455X (gfx1250) — compile-verified, matrix operand order checked by reading, not run on hardware
//
#include <hip/hip_runtime.h>
#include <hip/hip_bf16.h>

// ---------------- problem constants (from reference) ----------------
static constexpr int Bc = 16, Mq = 96, Kg = 32, Nn = 128;
static constexpr int Dd = 256, Cc = 128, DHd = 512, CHd = 256;
static constexpr int Hh = 8, DHe = 32, GEDc = 256;
static constexpr int NLc = 32, NFc = 16, ELc = 8, EFc = 8;
static constexpr long NNC = (long)Nn * Nn * Cc;      // per-batch edge slab

typedef __attribute__((ext_vector_type(16))) __bf16 v16bf;
typedef __attribute__((ext_vector_type(8)))  __bf16 v8bf;
typedef __attribute__((ext_vector_type(2)))  __bf16 v2bf;
typedef __attribute__((ext_vector_type(8)))  float  v8f;
typedef __attribute__((ext_vector_type(4)))  float  f4raw;   // plain vector (not HIP class)

// global-address-space loads (force global_load_* instead of flat_load_*)
__device__ __forceinline__ float ldg1(const float* p) {
  typedef const float __attribute__((address_space(1)))* gfp;
  return *(gfp)p;
}
__device__ __forceinline__ f4raw ldg4(const float* p) {
  typedef const f4raw __attribute__((address_space(1)))* gf4p;
  return *(gf4p)p;
}

// f32 -> bf16 RNE: let the backend pick the native cvt
__device__ __forceinline__ __bf16 f2bf(float f) { return (__bf16)f; }
__device__ __forceinline__ v2bf pk_bf16(float a, float b) {
#if __has_builtin(__builtin_amdgcn_cvt_pk_bf16_f32)
  return __builtin_amdgcn_cvt_pk_bf16_f32(a, b);
#else
  v2bf r; r[0] = f2bf(a); r[1] = f2bf(b); return r;
#endif
}
__device__ __forceinline__ void put2(v16bf& v, int i, float a, float b) {
  v2bf t = pk_bf16(a, b); v[i] = t[0]; v[i + 1] = t[1];
}
__device__ __forceinline__ float sigf(float x) { return 1.f / (1.f + __expf(-x)); }

// Per-lane 16-element A-operand slice. e<8 -> k=e ; e>=8 -> k=16+(e-8) (rel. to lane k-base)
template<bool S1>
__device__ __forceinline__ v16bf load16(const float* __restrict__ p, int sK) {
  v16bf r;
  if constexpr (S1) {
    const f4raw q0 = ldg4(p);
    const f4raw q1 = ldg4(p + 4);
    const f4raw q2 = ldg4(p + 16);
    const f4raw q3 = ldg4(p + 20);
    put2(r, 0,  q0[0], q0[1]); put2(r, 2,  q0[2], q0[3]);
    put2(r, 4,  q1[0], q1[1]); put2(r, 6,  q1[2], q1[3]);
    put2(r, 8,  q2[0], q2[1]); put2(r, 10, q2[2], q2[3]);
    put2(r, 12, q3[0], q3[1]); put2(r, 14, q3[2], q3[3]);
  } else {
#pragma unroll
    for (int e = 0; e < 8; e += 2)
      put2(r, e, ldg1(p + e * sK), ldg1(p + (e + 1) * sK));
#pragma unroll
    for (int e = 0; e < 8; e += 2)
      put2(r, 8 + e, ldg1(p + (16 + e) * sK), ldg1(p + (17 + e) * sK));
  }
  return r;
}

// ---------------- generic strided WMMA GEMM with LDS-staged B ----------------
// Block = 8 waves covering 8 M-tiles of ONE N-group (NT 16-wide subtiles).
// B tile (NT*16 x 32) is cooperatively converted to bf16 in LDS each K-step and
// reused by all 8 waves. Out(t,m,n) = epilogue(sum_k A(t,m,k)*Bt(t,n,k) + bias[n]).
// mode: 0 store, 1 +=, 2 relu store, 3 Out = sigmoid(acc)*Out, 4 Out += sigmoid(acc)*Aux
template<int NT, bool A1>
__global__ void __launch_bounds__(256)
gemm_wmma_t(const float* __restrict__ A, long aM, long aK, int aDiv, long aBo, long aBi,
            const float* __restrict__ Bt, long bN, long bK, int bDiv, long bBo, long bBi,
            const float* __restrict__ bias,
            float* __restrict__ Out, long oM, long oN, int oDiv, long oBo, long oBi,
            const float* __restrict__ Aux, long xM, long xN, int xDiv, long xBo, long xBi,
            int M, int Nc, int K, int mode)
{
  __shared__ alignas(16) __bf16 sB[NT * 16 * 32];
  const int tid = threadIdx.x;
  const int wave = tid >> 5, lane = tid & 31;
  const int Mtiles = M >> 4;
  const int MBlk = (Mtiles + 7) >> 3;
  const int tg = blockIdx.x / MBlk;              // N-group, shared by whole block
  const int mb = blockIdx.x % MBlk;
  const int tmRaw = mb * 8 + wave;
  const bool mValid = tmRaw < Mtiles;            // wave-uniform
  const int tm = mValid ? tmRaw : Mtiles - 1;    // clamp: keep all waves in barriers
  const int t = blockIdx.y;
  const long aOff = (long)(t / aDiv) * aBo + (long)(t % aDiv) * aBi;
  const long bOff = (long)(t / bDiv) * bBo + (long)(t % bDiv) * bBi;
  const long oOff = (long)(t / oDiv) * oBo + (long)(t % oDiv) * oBi;

  const int l15 = lane & 15, half = lane >> 4, kb = half * 8;
  const float* pA = A + aOff + (long)(tm * 16 + l15) * aM + (long)kb * aK;

  // cooperative B-fill mapping: each thread fills PER consecutive k of one n-row
  constexpr int PER = (NT * 16 * 32) / 256;      // 8 / 4 / 2
  constexpr int KGRP = 32 / PER;
  const int fn = tid / KGRP;                     // 0 .. NT*16-1
  const int fk = (tid % KGRP) * PER;
  const float* pBf = Bt + bOff + (long)(tg * NT * 16 + fn) * bN + (long)fk * bK;

  const v8f z = {0.f,0.f,0.f,0.f,0.f,0.f,0.f,0.f};
  v8f acc[NT];
#pragma unroll
  for (int u = 0; u < NT; ++u) acc[u] = z;

  for (int k0 = 0; k0 < K; k0 += 32) {
    __syncthreads();                             // previous consumers done
#pragma unroll
    for (int e = 0; e < PER; ++e)
      sB[fn * 32 + fk + e] = f2bf(ldg1(pBf + (long)e * bK));
    pBf += 32 * bK;
    __syncthreads();                             // slab visible

    const v16bf av = load16<A1>(pA, (int)aK);
    __builtin_prefetch(pA + 32 * aK, 0, 1);
    pA += 32 * aK;
#pragma unroll
    for (int u = 0; u < NT; ++u) {
      const v8bf lo = *(const v8bf*)(sB + (u * 16 + l15) * 32 + kb);
      const v8bf hi = *(const v8bf*)(sB + (u * 16 + l15) * 32 + kb + 16);
      const v16bf bv = __builtin_shufflevector(lo, hi,
          0,1,2,3,4,5,6,7,8,9,10,11,12,13,14,15);
      acc[u] = __builtin_amdgcn_wmma_f32_16x16x32_bf16(false, av, false, bv,
                                                       (short)0, acc[u], false, false);
    }
  }

  if (mValid) {
#pragma unroll
    for (int u = 0; u < NT; ++u) {
      const int n = (tg * NT + u) * 16 + l15;
      const float bb = bias ? bias[n] : 0.f;
#pragma unroll
      for (int r = 0; r < 8; ++r) {
        const int gm = tm * 16 + half * 8 + r;
        const float v = acc[u][r] + bb;
        const long oi = oOff + (long)gm * oM + (long)n * oN;
        if (mode == 0)      Out[oi] = v;
        else if (mode == 1) Out[oi] += v;
        else if (mode == 2) Out[oi] = v > 0.f ? v : 0.f;
        else if (mode == 3) Out[oi] *= sigf(v);
        else {
          const long xi = (long)(t / xDiv) * xBo + (long)(t % xDiv) * xBi
                        + (long)gm * xM + (long)n * xN;
          Out[oi] += sigf(v) * Aux[xi];
        }
      }
    }
  }
}

// E[m,:] += (oa[b,i,:] * ob[b,j,:]) @ W + bias, m=(b*Nn+i)*Nn+j ; NT=4, LDS-staged W
__global__ void __launch_bounds__(256)
gemm_outer_wmma(const float* __restrict__ oa, const float* __restrict__ ob,
                const float* __restrict__ W, const float* __restrict__ bias,
                float* __restrict__ E, long rowsTotal, int Kc, int Nc)
{
  constexpr int NT = 4;
  __shared__ alignas(16) __bf16 sB[NT * 16 * 32];
  const int tid = threadIdx.x;
  const int wave = tid >> 5, lane = tid & 31;
  const long Mtiles = rowsTotal >> 4;
  const long MBlk = (Mtiles + 7) >> 3;
  const int  tg = (int)(blockIdx.x / MBlk);
  const long mb = blockIdx.x % MBlk;
  const long tmRaw = mb * 8 + wave;
  const bool mValid = tmRaw < Mtiles;
  const long tm = mValid ? tmRaw : Mtiles - 1;
  const int l15 = lane & 15, half = lane >> 4, kb = half * 8;
  const long m  = tm * 16 + l15;
  const long b  = m / ((long)Nn * Nn);
  const long rm = m % ((long)Nn * Nn);
  const long i  = rm / Nn, j = rm % Nn;
  const float* pa = oa + (b * Nn + i) * Kc + kb;
  const float* pb = ob + (b * Nn + j) * Kc + kb;

  constexpr int PER = (NT * 16 * 32) / 256, KGRP = 32 / PER;
  const int fn = tid / KGRP, fk = (tid % KGRP) * PER;
  const float* pBf = W + (long)fk * Nc + tg * NT * 16 + fn;   // elem(n,k)=W[k*Nc+n]

  const v8f z = {0.f,0.f,0.f,0.f,0.f,0.f,0.f,0.f};
  v8f acc[NT];
#pragma unroll
  for (int u = 0; u < NT; ++u) acc[u] = z;

  for (int k0 = 0; k0 < Kc; k0 += 32) {
    __syncthreads();
#pragma unroll
    for (int e = 0; e < PER; ++e)
      sB[fn * 32 + fk + e] = f2bf(ldg1(pBf + (long)e * Nc));
    pBf += (long)32 * Nc;
    __syncthreads();

    v16bf av;
#pragma unroll
    for (int e = 0; e < 8; e += 2) {
      put2(av, e,     ldg1(pa + e) * ldg1(pb + e),
                      ldg1(pa + e + 1) * ldg1(pb + e + 1));
      put2(av, 8 + e, ldg1(pa + 16 + e) * ldg1(pb + 16 + e),
                      ldg1(pa + 17 + e) * ldg1(pb + 17 + e));
    }
    pa += 32; pb += 32;
#pragma unroll
    for (int u = 0; u < NT; ++u) {
      const v8bf lo = *(const v8bf*)(sB + (u * 16 + l15) * 32 + kb);
      const v8bf hi = *(const v8bf*)(sB + (u * 16 + l15) * 32 + kb + 16);
      const v16bf bv = __builtin_shufflevector(lo, hi,
          0,1,2,3,4,5,6,7,8,9,10,11,12,13,14,15);
      acc[u] = __builtin_amdgcn_wmma_f32_16x16x32_bf16(false, av, false, bv,
                                                       (short)0, acc[u], false, false);
    }
  }
  if (mValid) {
#pragma unroll
    for (int u = 0; u < NT; ++u) {
      const int n = (tg * NT + u) * 16 + l15;
      const float bb = bias[n];
#pragma unroll
      for (int r = 0; r < 8; ++r) {
        const long gm = tm * 16 + half * 8 + r;
        E[gm * Nc + n] += acc[u][r] + bb;
      }
    }
  }
}

// ---------------- wave-per-row LayerNorm ----------------
__global__ void __launch_bounds__(256)
ln_rows(const float* __restrict__ in, float* __restrict__ out,
        const float* __restrict__ g, const float* __restrict__ b, long rows, int d)
{
  const int wave = threadIdx.x >> 5, lane = threadIdx.x & 31;
  const long row = (long)blockIdx.x * 8 + wave;
  if (row >= rows) return;
  const float* x = in + row * d;
  float s = 0.f, s2 = 0.f;
  for (int i = lane; i < d; i += 32) { float v = ldg1(x + i); s += v; s2 += v * v; }
#pragma unroll
  for (int o = 16; o > 0; o >>= 1) { s += __shfl_xor(s, o, 32); s2 += __shfl_xor(s2, o, 32); }
  const float m = s / d;
  const float r = rsqrtf(s2 / d - m * m + 1e-5f);
  float* y = out + row * d;
  for (int i = lane; i < d; i += 32) y[i] = (ldg1(x + i) - m) * r * ldg1(g + i) + ldg1(b + i);
}

// ---------------- softmax over 128-wide rows, fused scale+bias ----------------
__global__ void __launch_bounds__(256)
k_softmax(float* __restrict__ sc, const float* __restrict__ bias, long rows, float scale)
{
  const int wave = threadIdx.x >> 5, lane = threadIdx.x & 31;
  const long row = (long)blockIdx.x * 8 + wave;
  if (row >= rows) return;
  float* p = sc + row * Nn; const float* bi = bias + row * Nn;
  float v[4]; float mx = -3.4e38f;
#pragma unroll
  for (int i = 0; i < 4; ++i) { v[i] = ldg1(p + lane + 32 * i) * scale + ldg1(bi + lane + 32 * i); mx = fmaxf(mx, v[i]); }
#pragma unroll
  for (int o = 16; o > 0; o >>= 1) mx = fmaxf(mx, __shfl_xor(mx, o, 32));
  float s = 0.f;
#pragma unroll
  for (int i = 0; i < 4; ++i) { v[i] = __expf(v[i] - mx); s += v[i]; }
#pragma unroll
  for (int o = 16; o > 0; o >>= 1) s += __shfl_xor(s, o, 32);
  const float inv = 1.f / s;
#pragma unroll
  for (int i = 0; i < 4; ++i) p[lane + 32 * i] = v[i] * inv;
}

// ---------------- bias einsum: out[b,h,i,j] = sum_c en[b,i,j,c]*W[c,h] ----------------
__global__ void __launch_bounds__(256)
k_bias_proj(const float* __restrict__ en, const float* __restrict__ w, float* __restrict__ out)
{
  const long total = (long)Bc * Hh * Nn * Nn;
  const long idx = (long)blockIdx.x * 256 + threadIdx.x;
  if (idx >= total) return;
  const int j = (int)(idx % Nn); long r = idx / Nn;
  const int i = (int)(r % Nn);   r /= Nn;
  const int h = (int)(r % Hh);
  const long b = r / Hh;
  const float* row = en + ((b * Nn + i) * Nn + j) * Cc;
  float s = 0.f;
  for (int c = 0; c < Cc; ++c) s += ldg1(row + c) * ldg1(w + c * Hh + h);
  out[idx] = s;
}

// ---------------- VALU GEMM for small N (1..8), flexible row mapping ----------------
__global__ void __launch_bounds__(256)
k_gemm_small(const float* __restrict__ A, int d1, int d2, long s0, long s1, long s2,
             const float* __restrict__ W, const float* __restrict__ bias,
             float* __restrict__ out, long rows, int Nc, int K)
{
  const long idx = (long)blockIdx.x * 256 + threadIdx.x;
  if (idx >= rows * (long)Nc) return;
  const long m = idx / Nc; const int n = (int)(idx % Nc);
  const long p = m / ((long)d1 * d2);
  const long q = (m / d2) % d1;
  const long r = m % d2;
  const float* a = A + p * s0 + q * s1 + r * s2;
  float acc = bias ? ldg1(bias + n) : 0.f;
  for (int k = 0; k < K; ++k) acc += ldg1(a + k) * ldg1(W + (long)k * Nc + n);
  out[idx] = acc;
}

// ---------------- init kernels ----------------
__global__ void __launch_bounds__(256)
k_build_X(const float* __restrict__ qn, const float* __restrict__ xl, float* __restrict__ X)
{
  const long total = (long)Bc * Nn * Dd;
  const long idx = (long)blockIdx.x * 256 + threadIdx.x;
  if (idx >= total) return;
  const int d = (int)(idx % Dd); const long r = idx / Dd;
  const int i = (int)(r % Nn);   const long b = r / Nn;
  X[idx] = (i < Mq) ? ldg1(qn + (long)i * Dd + d)
                    : ldg1(xl + ((b * Kg) + (i - Mq)) * Dd + d);
}
__global__ void __launch_bounds__(256)
k_bcast_E(const float* __restrict__ qe, float* __restrict__ E)
{
  const long total = (long)Bc * NNC;
  const long idx = (long)blockIdx.x * 256 + threadIdx.x;
  if (idx >= total) return;
  E[idx] = ldg1(qe + idx % NNC);
}

// ---------------- host-side launch helpers ----------------
template<int NT>
static inline void launch_nt(dim3 grid, hipStream_t st, bool a1,
                             const float* A, long aM, long aK, int aDiv, long aBo, long aBi,
                             const float* Bt, long bN, long bK, int bDiv, long bBo, long bBi,
                             const float* bias,
                             float* Out, long oM, long oN, int oDiv, long oBo, long oBi,
                             const float* Aux, long xM, long xN, int xDiv, long xBo, long xBi,
                             int M, int Nc, int K, int mode)
{
#define GO(AA) gemm_wmma_t<NT, AA><<<grid, 256, 0, st>>>( \
      A, aM, aK, aDiv, aBo, aBi, Bt, bN, bK, bDiv, bBo, bBi, bias, \
      Out, oM, oN, oDiv, oBo, oBi, Aux, xM, xN, xDiv, xBo, xBi, M, Nc, K, mode)
  if (a1) GO(true); else GO(false);
#undef GO
}

static inline void G(hipStream_t st, int nb, int M, int Nc, int K,
                     const float* A, long aM, long aK, int aDiv, long aBo, long aBi,
                     const float* Bt, long bN, long bK, int bDiv, long bBo, long bBi,
                     const float* bias,
                     float* Out, long oM, long oN, int oDiv, long oBo, long oBi,
                     const float* Aux, long xM, long xN, int xDiv, long xBo, long xBi,
                     int mode)
{
  const int NT = (Nc % 64 == 0) ? 4 : (Nc % 32 == 0) ? 2 : 1;
  const int Mtiles = M / 16, MBlk = (Mtiles + 7) / 8, NtG = Nc / (16 * NT);
  dim3 grid((unsigned)(MBlk * NtG), (unsigned)nb);
  const bool a1 = (aK == 1);
  if (NT == 4)      launch_nt<4>(grid, st, a1, A, aM, aK, aDiv, aBo, aBi, Bt, bN, bK, bDiv, bBo, bBi, bias, Out, oM, oN, oDiv, oBo, oBi, Aux, xM, xN, xDiv, xBo, xBi, M, Nc, K, mode);
  else if (NT == 2) launch_nt<2>(grid, st, a1, A, aM, aK, aDiv, aBo, aBi, Bt, bN, bK, bDiv, bBo, bBi, bias, Out, oM, oN, oDiv, oBo, oBi, Aux, xM, xN, xDiv, xBo, xBi, M, Nc, K, mode);
  else              launch_nt<1>(grid, st, a1, A, aM, aK, aDiv, aBo, aBi, Bt, bN, bK, bDiv, bBo, bBi, bias, Out, oM, oN, oDiv, oBo, oBi, Aux, xM, xN, xDiv, xBo, xBi, M, Nc, K, mode);
}
static inline void GW(hipStream_t st, long rows, int Nc, int K,
                      const float* A, long lda, const float* W, const float* bias,
                      float* Out, long ldo, int mode,
                      const float* Aux = nullptr, long ldx = 1)
{
  G(st, 1, (int)rows, Nc, K, A, lda, 1, 1, 0, 0, W, 1, Nc, 1, 0, 0, bias,
    Out, ldo, 1, 1, 0, 0, Aux, ldx, 1, 1, 0, 0, mode);
}
static inline void LN(hipStream_t st, const float* in, float* out,
                      const float* g, const float* b, long rows, int d)
{
  ln_rows<<<(unsigned)((rows + 7) / 8), 256, 0, st>>>(in, out, g, b, rows, d);
}

extern "C" void kernel_launch(void* const* d_in, const int* in_sizes, int n_in,
                              void* d_out, int out_size, void* d_ws, size_t ws_size,
                              hipStream_t stream)
{
  (void)in_sizes; (void)out_size; (void)ws_size;
  if (n_in < 223) return;
  auto F = [&](int i) { return (const float*)d_in[i]; };
  const int LB = 7, LS = 49;
  auto LP = [&](int l, int j) { return (const float*)d_in[LB + l * LS + j]; };
  const int hb = LB + 4 * LS; // 203

  // ---- workspace layout (floats) ----
  float* ws = (float*)d_ws; size_t off = 0;
  auto alloc = [&](size_t n) { float* p = ws + off; off += n; return p; };
  float* E    = alloc((size_t)Bc * NNC);
  float* EN   = alloc((size_t)Bc * NNC);
  float* X    = alloc((size_t)Bc * Nn * Dd);
  float* XN   = alloc((size_t)Bc * Nn * Dd);
  float* Q    = alloc((size_t)Bc * Nn * Dd);
  float* Kb   = alloc((size_t)Bc * Nn * Dd);
  float* V    = alloc((size_t)Bc * Nn * Dd);
  float* O    = alloc((size_t)Bc * Nn * Dd);
  float* SC   = alloc((size_t)Bc * Hh * Nn * Nn);
  float* BIAS = alloc((size_t)Bc * Hh * Nn * Nn);
  float* HX   = alloc((size_t)Bc * Nn * DHd);
  float* OA   = alloc((size_t)Bc * Nn * Cc);
  float* OB   = alloc((size_t)Bc * Nn * Cc);
  float* TA   = alloc((size_t)NNC);
  float* TB   = alloc((size_t)NNC);
  float* T    = alloc((size_t)NNC);
  float* U    = alloc((size_t)NNC);
  float* HE   = alloc((size_t)Nn * Nn * CHd);
  float* XLAT = alloc((size_t)Bc * Kg * Dd);
  float* HG   = alloc((size_t)Bc * Kg * DHd);
  float* HQ2  = alloc((size_t)Bc * Mq * DHd);
  float* HQE  = alloc((size_t)Mq * Mq * CHd);

  float* out  = (float*)d_out;
  float* o_ne = out;
  float* o_h  = out + 393216;
  float* o_nl = out + 394752;
  float* o_nf = out + 443904;
  float* o_A  = out + 468480;
  float* o_el = out + 615936;
  float* o_ef = out + 1795584;

  const long BN = (long)Bc * Nn;            // 2048
  const long BNN = (long)Bc * Nn * Nn;      // 262144
  const long NNr = (long)Nn * Nn;           // 16384

  // ---- stage 0: graph MLP + assemble X, broadcast E ----
  GW(stream, (long)Bc * Kg, DHd, GEDc, F(0), GEDc, F(3), F(4), HG, DHd, 2);
  GW(stream, (long)Bc * Kg, Dd, DHd, HG, DHd, F(5), F(6), XLAT, Dd, 0);
  k_build_X<<<(unsigned)(((long)Bc * Nn * Dd + 255) / 256), 256, 0, stream>>>(F(1), XLAT, X);
  k_bcast_E<<<(unsigned)(((long)Bc * NNC + 255) / 256), 256, 0, stream>>>(F(2), E);

  // ---- layers ----
  for (int l = 0; l < 4; ++l) {
    LN(stream, X, XN, LP(l, 0), LP(l, 1), BN, Dd);
    GW(stream, BN, Dd, Dd, XN, Dd, LP(l, 5), LP(l, 6), Q, Dd, 0);
    GW(stream, BN, Dd, Dd, XN, Dd, LP(l, 7), LP(l, 8), Kb, Dd, 0);
    GW(stream, BN, Dd, Dd, XN, Dd, LP(l, 9), LP(l, 10), V, Dd, 0);
    LN(stream, E, EN, LP(l, 2), LP(l, 3), BNN, Cc);
    k_bias_proj<<<(unsigned)(((long)Bc * Hh * NNr + 255) / 256), 256, 0, stream>>>(EN, LP(l, 4), BIAS);
    // scores[b,h] = q @ k^T
    G(stream, Bc * Hh, Nn, Nn, DHe,
      Q, Dd, 1, Hh, (long)Nn * Dd, DHe,
      Kb, Dd, 1, Hh, (long)Nn * Dd, DHe,
      nullptr,
      SC, Nn, 1, 1, NNr, 0,
      nullptr, 0, 1, 1, 0, 0, 0);
    k_softmax<<<(unsigned)(((long)Bc * Hh * Nn + 7) / 8), 256, 0, stream>>>(
        SC, BIAS, (long)Bc * Hh * Nn, 0.17677669529663687f);
    // o[b,h] = probs @ v
    G(stream, Bc * Hh, Nn, DHe, Nn,
      SC, Nn, 1, 1, NNr, 0,
      V, 1, Dd, Hh, (long)Nn * Dd, DHe,
      nullptr,
      O, Dd, 1, Hh, (long)Nn * Dd, DHe,
      nullptr, 0, 1, 1, 0, 0, 0);
    GW(stream, BN, Dd, Dd, O, Dd, LP(l, 11), LP(l, 12), X, Dd, 1);
    LN(stream, X, XN, LP(l, 13), LP(l, 14), BN, Dd);
    GW(stream, BN, DHd, Dd, XN, Dd, LP(l, 15), LP(l, 16), HX, DHd, 2);
    GW(stream, BN, Dd, DHd, HX, DHd, LP(l, 17), LP(l, 18), X, Dd, 1);
    LN(stream, X, XN, LP(l, 19), LP(l, 20), BN, Dd);
    GW(stream, BN, Cc, Dd, XN, Dd, LP(l, 21), LP(l, 22), OA, Cc, 0);
    GW(stream, BN, Cc, Dd, XN, Dd, LP(l, 23), LP(l, 24), OB, Cc, 0);
    {
      const long MBlk = (BNN / 16 + 7) / 8, NtG = Cc / 64;
      gemm_outer_wmma<<<(unsigned)(MBlk * NtG), 256, 0, stream>>>(
          OA, OB, LP(l, 25), LP(l, 26), E, BNN, Cc, Cc);
    }
    LN(stream, E, EN, LP(l, 27), LP(l, 28), BNN, Cc);
    for (int b = 0; b < Bc; ++b) {
      const float* et = EN + (long)b * NNC;
      float* Eb = E + (long)b * NNC;
      GW(stream, NNr, Cc, Cc, et, Cc, LP(l, 29), LP(l, 30), TA, Cc, 0);
      GW(stream, NNr, Cc, Cc, et, Cc, LP(l, 31), LP(l, 32), TA, Cc, 3);
      GW(stream, NNr, Cc, Cc, et, Cc, LP(l, 33), LP(l, 34), TB, Cc, 0);
      GW(stream, NNr, Cc, Cc, et, Cc, LP(l, 35), LP(l, 36), TB, Cc, 3);
      G(stream, Cc, Nn, Nn, Nn,
        TA, (long)Nn * Cc, Cc, 1, 1, 0,
        TB, (long)Nn * Cc, Cc, 1, 1, 0,
        nullptr,
        T, (long)Nn * Cc, Cc, 1, 1, 0,
        nullptr, 0, 1, 1, 0, 0, 0);
      LN(stream, T, T, LP(l, 37), LP(l, 38), NNr, Cc);
      GW(stream, NNr, Cc, Cc, T, Cc, LP(l, 39), LP(l, 40), U, Cc, 0);
      G(stream, 1, (int)NNr, Cc, Cc,
        et, Cc, 1, 1, 0, 0,
        LP(l, 41), 1, Cc, 1, 0, 0,
        LP(l, 42),
        Eb, Cc, 1, 1, 0, 0,
        U, Cc, 1, 1, 0, 0, 4);
    }
    LN(stream, E, EN, LP(l, 43), LP(l, 44), BNN, Cc);
    for (int b = 0; b < Bc; ++b) {
      GW(stream, NNr, CHd, Cc, EN + (long)b * NNC, Cc, LP(l, 45), LP(l, 46), HE, CHd, 2);
      GW(stream, NNr, Cc, CHd, HE, CHd, LP(l, 47), LP(l, 48), E + (long)b * NNC, Cc, 1);
    }
  }

  // ---- heads ----
  G(stream, Bc, Mq, DHd, Dd,
    X, Dd, 1, 1, (long)Nn * Dd, 0,
    F(hb + 2), 1, DHd, 1, 0, 0, F(hb + 3),
    HQ2, DHd, 1, 1, (long)Mq * DHd, 0,
    nullptr, 0, 1, 1, 0, 0, 2);
  GW(stream, (long)Bc * Mq, Dd, DHd, HQ2, DHd, F(hb + 4), F(hb + 5), o_ne, Dd, 0);
  k_gemm_small<<<(unsigned)(((long)Bc * Mq + 255) / 256), 256, 0, stream>>>(
      X, 1, Mq, (long)Nn * Dd, 0, Dd, F(hb + 0), F(hb + 1), o_h, (long)Bc * Mq, 1, Dd);
  G(stream, Bc, Mq, NLc, Dd,
    X, Dd, 1, 1, (long)Nn * Dd, 0,
    F(hb + 6), 1, NLc, 1, 0, 0, F(hb + 7),
    o_nl, NLc, 1, 1, (long)Mq * NLc, 0,
    nullptr, 0, 1, 1, 0, 0, 0);
  G(stream, Bc, Mq, DHd, Dd,
    X, Dd, 1, 1, (long)Nn * Dd, 0,
    F(hb + 8), 1, DHd, 1, 0, 0, F(hb + 9),
    HQ2, DHd, 1, 1, (long)Mq * DHd, 0,
    nullptr, 0, 1, 1, 0, 0, 2);
  GW(stream, (long)Bc * Mq, NFc, DHd, HQ2, DHd, F(hb + 10), F(hb + 11), o_nf, NFc, 0);
  {
    const long rows = (long)Bc * Mq * Mq;
    k_gemm_small<<<(unsigned)((rows + 255) / 256), 256, 0, stream>>>(
        E, Mq, Mq, (long)Nn * Nn * Cc, (long)Nn * Cc, Cc,
        F(hb + 12), F(hb + 13), o_A, rows, 1, Cc);
    k_gemm_small<<<(unsigned)((rows * ELc + 255) / 256), 256, 0, stream>>>(
        E, Mq, Mq, (long)Nn * Nn * Cc, (long)Nn * Cc, Cc,
        F(hb + 14), F(hb + 15), o_el, rows, ELc, Cc);
  }
  for (int b = 0; b < Bc; ++b) {
    G(stream, Mq, Mq, CHd, Cc,
      E + (long)b * NNC, Cc, 1, 1, (long)Nn * Cc, 0,
      F(hb + 16), 1, CHd, 1, 0, 0, F(hb + 17),
      HQE, CHd, 1, 1, (long)Mq * CHd, 0,
      nullptr, 0, 1, 1, 0, 0, 2);
    const long rows = (long)Mq * Mq;
    k_gemm_small<<<(unsigned)((rows * EFc + 255) / 256), 256, 0, stream>>>(
        HQE, 1, 1, CHd, 0, 0, F(hb + 18), F(hb + 19),
        o_ef + (long)b * rows * EFc, rows, EFc, CHd);
  }
}
